// Hysteresis_8486855377050
// MI455X (gfx1250) — matrix operand order repaired and checked by reading, not compiled, not run on hardware
//
#include <hip/hip_runtime.h>
#include <math.h>

// ---------------------------------------------------------------------------
// Preisach hysteresis on gfx1250 (MI455X, wave32, WMMA).
//   phase 0: normalize field history, softplus densities (+hi/lo f16 split),
//            total density sum.
//   phase 1: per-relay time scan; every 16 steps contract states x dens via
//            v_wmma_f32_16x16x32_f16 (dens split hi+lo for f32 accuracy).
//   phase 2: deterministic 128-way reduction + scale by B_SAT/total.
// ---------------------------------------------------------------------------

typedef _Float16 half_t;
typedef __attribute__((ext_vector_type(16))) _Float16 v16h;
typedef __attribute__((ext_vector_type(8)))  _Float16 v8h;
typedef __attribute__((ext_vector_type(8)))  float    v8f;

#define NMESH 256
#define MSTEP 4000
#define NREL  (NMESH * (NMESH + 1) / 2)   // 32896 = 128.5 * 256
#define TILE  16
#define NTILE (MSTEP / TILE)              // 250 exactly
#define WGSZ  256
#define NWG   (NREL / WGSZ)               // 128

// workspace byte offsets
#define OFF_TOTAL 0
#define OFF_HS    256                      // 4001 f32
#define OFF_DHI   16384                    // 32896 f16 (32B aligned)
#define OFF_DLO   82432                    // 32896 f16 (32B aligned)
#define OFF_PART  148480                   // 128 * 4000 f32

// ---------------------------------------------------------------------------
// Phase 0: field normalization + softplus densities + total (single WG).
// ---------------------------------------------------------------------------
__global__ __launch_bounds__(256)
void hyst_prep(const float* __restrict__ h_data,
               const float* __restrict__ raw,
               float* __restrict__ hs,
               half_t* __restrict__ dhi,
               half_t* __restrict__ dlo,
               float* __restrict__ total) {
  const int tid = threadIdx.x;
  if (tid == 0) hs[0] = 1.0f;                       // -H_MIN
  for (int i = tid; i < MSTEP; i += 256)
    hs[1 + i] = (h_data[i] + 1.0f) * 0.5f;          // (h - H_MIN)/(H_MAX - H_MIN)

  double lsum = 0.0;
  for (int f = tid; f < NREL; f += 256) {
    float x = raw[f];
    float d = fmaxf(x, 0.0f) + log1pf(expf(-fabsf(x)));   // softplus
    half_t hi = (half_t)d;
    dhi[f] = hi;
    dlo[f] = (half_t)(d - (float)hi);
    lsum += (double)d;
  }
  __shared__ double red[256];
  red[tid] = lsum;
  __syncthreads();
  for (int s = 128; s > 0; s >>= 1) {
    if (tid < s) red[tid] += red[tid + s];
    __syncthreads();
  }
  if (tid == 0) *total = (float)red[0];
}

// ---------------------------------------------------------------------------
// Phase 1: scan + WMMA contraction.
//   WG = 256 threads = 8 waves; wave w owns 32 relays (A columns K=0..31).
//   Per 16-step tile: states -> LDS -> A fragment, D = A*Bhi + A*Blo,
//   wave partials combined via LDS, written to partial[wg][t].
// ---------------------------------------------------------------------------
__global__ __launch_bounds__(256)
void hyst_main(const float* __restrict__ hs_g,
               const half_t* __restrict__ dhi,
               const half_t* __restrict__ dlo,
               float* __restrict__ partial) {
  __shared__ float hs_s[MSTEP + 1];                       // 16004 B
  __shared__ __align__(16) half_t atile[8][TILE][40];     // padded stride, 10240 B
  __shared__ float wbuf[8][TILE];                         // 512 B

  const int tid  = threadIdx.x;
  const int wg   = blockIdx.x;
  const int w    = tid >> 5;
  const int lane = tid & 31;

  for (int i = tid; i <= MSTEP; i += 256) hs_s[i] = hs_g[i];
  __syncthreads();

  // flat tril index -> (i=alpha row, j=beta col)
  const int f = wg * WGSZ + tid;
  int ri = (int)((sqrtf(8.0f * (float)f + 1.0f) - 1.0f) * 0.5f);
  while (ri * (ri + 1) / 2 > f) --ri;
  while ((ri + 1) * (ri + 2) / 2 <= f) ++ri;
  const int rj = f - ri * (ri + 1) / 2;
  const float alpha = (ri == NMESH - 1) ? 1.0f : (float)ri * (1.0f / (NMESH - 1));
  const float beta  = (rj == NMESH - 1) ? 1.0f : (float)rj * (1.0f / (NMESH - 1));

  // B fragments (constant over tiles): B[k][n] = dens_k for all n (replicated
  // columns). K order used here (same as A): lanes 0-15 hold K=0..15, lanes 16-31
  // hold K=16..31, packed pairs per VGPR -> contiguous 16-half load.
  const int boff = wg * WGSZ + w * 32 + ((lane >> 4) << 4);
  const v16h bhi = *(const v16h*)(dhi + boff);
  const v16h blo = *(const v16h*)(dlo + boff);

  // scan step tau = 0 (h = 1.0, p = hs[4000])
  float state = -1.0f;
  float hprev = hs_s[MSTEP];
  {
    float h = hs_s[0];
    state = (h > hprev) ? ((alpha < h) ? 1.0f : state)
          : ((h < hprev) ? ((beta > h) ? -1.0f : state) : state);
    hprev = h;
  }

  for (int tt = 0; tt < NTILE; ++tt) {
    const int tau0 = 1 + tt * TILE;
#pragma unroll
    for (int r = 0; r < TILE; ++r) {
      float h = hs_s[tau0 + r];
      state = (h > hprev) ? ((alpha < h) ? 1.0f : state)
            : ((h < hprev) ? ((beta > h) ? -1.0f : state) : state);
      hprev = h;
      atile[w][r][lane] = (half_t)state;                  // [time][relay]
    }
    // cross-lane RAW through LDS within this wave: wait DS ops complete
    asm volatile("s_wait_dscnt 0" ::: "memory");

    // A fragment (16-bit A 16x32): lane m = time row m;
    // lanes 0-15: K=0..15; lanes 16-31: K=16..31
    const int row  = lane & 15;
    const int koff = (lane >> 4) << 4;                    // 0 or 16
    v8h lo8 = *(const v8h*)(&atile[w][row][koff]);
    v8h hi8 = *(const v8h*)(&atile[w][row][8 + koff]);
    v16h A = __builtin_shufflevector(lo8, hi8,
                                     0, 1, 2, 3, 4, 5, 6, 7,
                                     8, 9, 10, 11, 12, 13, 14, 15);

    v8f c = {};
    c = __builtin_amdgcn_wmma_f32_16x16x32_f16(false, A, false, bhi,
                                               (short)0, c, false, false);
    c = __builtin_amdgcn_wmma_f32_16x16x32_f16(false, A, false, blo,
                                               (short)0, c, false, false);

    // D column 0: lane 0 -> time rows 0-7, lane 16 -> time rows 8-15
    if (row == 0) {
      const int rbase = (lane >> 4) << 3;
#pragma unroll
      for (int v = 0; v < 8; ++v) wbuf[w][rbase + v] = c[v];
    }
    __syncthreads();
    if (tid < TILE) {
      float s = 0.0f;
#pragma unroll
      for (int ww = 0; ww < 8; ++ww) s += wbuf[ww][tid];
      partial[(size_t)wg * MSTEP + tt * TILE + tid] = s;
    }
    __syncthreads();
  }
}

// ---------------------------------------------------------------------------
// Phase 2: deterministic reduction over 128 workgroup partials + scale.
// ---------------------------------------------------------------------------
__global__ __launch_bounds__(256)
void hyst_reduce(const float* __restrict__ partial,
                 const float* __restrict__ total,
                 float* __restrict__ out) {
  const int t = blockIdx.x * 256 + threadIdx.x;
  if (t >= MSTEP) return;
  float s = 0.0f;
  for (int g = 0; g < NWG; ++g) s += partial[(size_t)g * MSTEP + t];
  out[t] = s * (1.0f / *total);                           // B_SAT = 1
}

// ---------------------------------------------------------------------------
extern "C" void kernel_launch(void* const* d_in, const int* in_sizes, int n_in,
                              void* d_out, int out_size, void* d_ws, size_t ws_size,
                              hipStream_t stream) {
  const float* h_data = (const float*)d_in[0];
  const float* raw    = (const float*)d_in[1];

  char* ws = (char*)d_ws;
  float*  total   = (float*)(ws + OFF_TOTAL);
  float*  hs      = (float*)(ws + OFF_HS);
  half_t* dhi     = (half_t*)(ws + OFF_DHI);
  half_t* dlo     = (half_t*)(ws + OFF_DLO);
  float*  partial = (float*)(ws + OFF_PART);

  hyst_prep<<<1, 256, 0, stream>>>(h_data, raw, hs, dhi, dlo, total);
  hyst_main<<<NWG, 256, 0, stream>>>(hs, dhi, dlo, partial);
  hyst_reduce<<<(MSTEP + 255) / 256, 256, 0, stream>>>(partial, total,
                                                       (float*)d_out);
}
